// LoopVisibleLSTM_38414187495768
// MI455X (gfx1250) — matrix operand order repaired and checked by reading, not compiled, not run on hardware
//
#include <hip/hip_runtime.h>
#include <hip/hip_bf16.h>
#include <cstddef>
#include <cstdint>

// ---------------------------------------------------------------------------
// LSTM on MI455X (gfx1250, wave32, WMMA).
//   Phase 1: x_proj = x @ W_ih^T + b_ih + b_hh   (bf16 WMMA, f32 accum)
//   Phase 2: persistent kernel, 1024 serial steps; W_hh AND h resident in LDS
//            (h staged per step with global_load_async_to_lds_b128),
//            cell state c in registers, grid-wide atomic barrier per step.
// ---------------------------------------------------------------------------

typedef __bf16 bf16_t;
typedef bf16_t v16bf __attribute__((ext_vector_type(16)));
typedef bf16_t v8bf  __attribute__((ext_vector_type(8)));
typedef float  v8f   __attribute__((ext_vector_type(8)));

#define T_STEPS 1024
#define BATCH   64
#define ISZ     1024
#define HSZ     1024
#define G4      4096
#define NBLK    64            // recurrent blocks = HSZ/16
#define WSTRIDE 1032          // padded bf16 row stride in LDS (bank spread)

// Build a 16x16x32 bf16 WMMA operand from two 16-byte chunks.
__device__ __forceinline__ v16bf ld16(const bf16_t* lo, const bf16_t* hi) {
  v8bf a = *(const v8bf*)lo;
  v8bf b = *(const v8bf*)hi;
  return __builtin_shufflevector(a, b, 0,1,2,3,4,5,6,7,8,9,10,11,12,13,14,15);
}

__device__ __forceinline__ v8f wmma_bf16(v16bf a, v16bf b, v8f c) {
  return __builtin_amdgcn_wmma_f32_16x16x32_bf16(false, a, false, b,
                                                 (short)0, c, false, false);
}

__device__ __forceinline__ float sigmoidf_fast(float x) {
  return 1.0f / (1.0f + __expf(-x));
}

// CDNA5 async copy: 16 bytes global -> LDS per lane, tracked on ASYNCcnt.
__device__ __forceinline__ void async_ld_b128(unsigned lds_byte_addr,
                                              const void* gaddr) {
  asm volatile("global_load_async_to_lds_b128 %0, %1, off"
               :: "v"(lds_byte_addr), "v"(gaddr) : "memory");
}
__device__ __forceinline__ void wait_asynccnt0() {
  asm volatile("s_wait_asynccnt 0x0" ::: "memory");
}

// ---------------------------------------------------------------------------
// init: zero grid-barrier counter, convert h0 (f32) -> bf16 double buffer 0
// ---------------------------------------------------------------------------
__global__ void lstm_init(const float* __restrict__ h0,
                          bf16_t* __restrict__ hbuf,
                          unsigned* __restrict__ counter) {
  int i = blockIdx.x * blockDim.x + threadIdx.x;
  if (i == 0) *counter = 0u;
  if (i < BATCH * HSZ) hbuf[i] = (bf16_t)h0[i];
}

// ---------------------------------------------------------------------------
// Phase 1: x_proj[m, n] = sum_k x[m,k] * W_ih[n,k] + b_ih[n] + b_hh[n]
// 64x64 block tile, 8 waves x 2 tiles (shared A frag), K staged 32 at a time,
// f32 -> bf16 conversion fused into the staging loads.
// ---------------------------------------------------------------------------
__global__ __launch_bounds__(256) void gemm_xproj(
    const float* __restrict__ x, const float* __restrict__ W,
    const float* __restrict__ b_ih, const float* __restrict__ b_hh,
    float* __restrict__ xp) {
  __shared__ bf16_t A_sh[64 * 40];   // stride 40 -> conflict-free b128 reads
  __shared__ bf16_t B_sh[64 * 40];

  const int tid  = threadIdx.x;
  const int lane = tid & 31;
  const int wave = tid >> 5;
  const int r  = wave & 3;           // M tile 0..3
  const int cp = wave >> 2;          // N tiles cp and cp+2
  const size_t m0 = (size_t)blockIdx.y * 64;
  const size_t n0 = (size_t)blockIdx.x * 64;

  const int srow = tid >> 2;         // staging: row 0..63
  const int sseg = (tid & 3) * 8;    // staging: 8 elems per thread

  const int m   = lane & 15;
  const int ahi = (lane >> 4) * 16;  // A frag K sub-offset
  const int bhi = (lane >> 4) * 16;  // B frag K sub-offset
  const bf16_t* arow  = A_sh + (r  * 16 + m) * 40;
  const bf16_t* b0row = B_sh + (cp * 16 + m) * 40 + bhi;
  const bf16_t* b1row = B_sh + ((cp + 2) * 16 + m) * 40 + bhi;

  v8f acc0 = {}; v8f acc1 = {};

  for (int k0 = 0; k0 < ISZ; k0 += 32) {
    const float* ap = x + (m0 + srow) * ISZ + k0 + sseg;
    const float* bp = W + (n0 + srow) * ISZ + k0 + sseg;
    bf16_t* ad = A_sh + srow * 40 + sseg;
    bf16_t* bd = B_sh + srow * 40 + sseg;
#pragma unroll
    for (int u = 0; u < 8; u++) { ad[u] = (bf16_t)ap[u]; bd[u] = (bf16_t)bp[u]; }
    __syncthreads();

    v16bf a  = ld16(arow + ahi, arow + ahi + 8);
    v16bf b0 = ld16(b0row, b0row + 8);
    v16bf b1 = ld16(b1row, b1row + 8);
    acc0 = wmma_bf16(a, b0, acc0);
    acc1 = wmma_bf16(a, b1, acc1);
    __syncthreads();
  }

  // epilogue: C/D layout -> element v is row (v + 8*(lane>>4)), col (lane&15)
  const int n     = lane & 15;
  const int rbase = r * 16 + (lane >> 4) * 8;
  const size_t col0 = n0 + cp * 16 + n;
  const size_t col1 = n0 + (cp + 2) * 16 + n;
  const float bias0 = b_ih[col0] + b_hh[col0];
  const float bias1 = b_ih[col1] + b_hh[col1];
#pragma unroll
  for (int v = 0; v < 8; v++) {
    size_t row = m0 + rbase + v;
    xp[row * G4 + col0] = acc0[v] + bias0;
    xp[row * G4 + col1] = acc1[v] + bias1;
  }
}

// ---------------------------------------------------------------------------
// Phase 2: persistent recurrence. Block j owns hidden cols [16j,16j+16) for
// all 4 gates. W_hh (64 rows x 1024, bf16) stays in LDS for all 1024 steps;
// h (64 x 1024, bf16) is staged into LDS per step with async-to-LDS copies
// from a global double buffer. c lives in VGPRs.
// LDS: 129 KB (W) + 129 KB (h) + 16 KB (gates) = 274 KB <= 320 KB/WG.
// ---------------------------------------------------------------------------
__global__ __launch_bounds__(256) void lstm_recurrent(
    const float* __restrict__ W_hh, const float* __restrict__ xp,
    const float* __restrict__ c0, float* __restrict__ out,
    bf16_t* __restrict__ hbuf, unsigned* __restrict__ counter) {
  extern __shared__ char smem[];
  bf16_t* W_sh = (bf16_t*)smem;                                   // 64*WSTRIDE
  bf16_t* h_sh = (bf16_t*)(smem + 64 * WSTRIDE * sizeof(bf16_t)); // 64*WSTRIDE
  float*  g_sh = (float*)(smem + 2 * 64 * WSTRIDE * sizeof(bf16_t)); // [4][64][16]
  const unsigned h_base = (unsigned)(size_t)(void*)h_sh;  // LDS byte offset

  const int tid  = threadIdx.x;
  const int lane = tid & 31;
  const int wave = tid >> 5;
  const int r  = wave & 3;         // batch row tile (16 rows)
  const int gp = wave >> 2;        // gates gp and gp+2
  const int j0 = blockIdx.x * 16;

  // Stage W_hh rows {g*1024 + j0 + jj} -> W_sh[g*16+jj][:], f32 -> bf16.
  for (int i = tid; i < 64 * (HSZ / 4); i += 256) {
    int lr   = i >> 8;                       // local row 0..63
    int seg  = (i & 255) * 4;                // 4 f32 per iteration
    int grow = ((lr >> 4) << 10) + j0 + (lr & 15);
    const float4 w4 = *(const float4*)(W_hh + (size_t)grow * HSZ + seg);
    bf16_t* dst = W_sh + lr * WSTRIDE + seg;
    dst[0] = (bf16_t)w4.x; dst[1] = (bf16_t)w4.y;
    dst[2] = (bf16_t)w4.z; dst[3] = (bf16_t)w4.w;
  }

  // Per-thread cell state: fixed (b, jj) mapping, c kept in registers.
  float c_reg[4];
  int bb[4], jj[4];
#pragma unroll
  for (int q = 0; q < 4; q++) {
    int p = tid + q * 256;
    bb[q] = p >> 4; jj[q] = p & 15;
    c_reg[q] = c0[bb[q] * HSZ + j0 + jj[q]];
  }
  __syncthreads();

  const int m   = lane & 15;
  const int ahi = (lane >> 4) * 16;
  const bf16_t* hrow  = h_sh + (r * 16 + m) * WSTRIDE;
  const bf16_t* b0row = W_sh + (gp * 16 + m) * WSTRIDE + (lane >> 4) * 16;
  const bf16_t* b1row = W_sh + ((gp + 2) * 16 + m) * WSTRIDE + (lane >> 4) * 16;
  const int n     = lane & 15;
  const int rbase = r * 16 + (lane >> 4) * 8;

  for (int t = 0; t < T_STEPS; t++) {
    const bf16_t* hread  = hbuf + (size_t)(t & 1) * (BATCH * HSZ);
    bf16_t*       hwrite = hbuf + (size_t)((t + 1) & 1) * (BATCH * HSZ);

    // Async-stage h (128 KB) into LDS: 16 B per lane per issue, ASYNCcnt.
#pragma unroll 4
    for (int i = 0; i < 32; i++) {
      int flat = tid + i * 256;
      int row = flat >> 7;                   // 128 x 16B segments per row
      int seg = flat & 127;
      async_ld_b128(h_base + (unsigned)(row * (WSTRIDE * 2) + seg * 16),
                    hread + (size_t)row * HSZ + seg * 8);
    }
    // Warm L2/L0 for this step's x_proj slice while the GEMM runs.
    const float* xpt = xp + (size_t)t * BATCH * G4;
#pragma unroll
    for (int q = 0; q < 4; q++) {
      __builtin_prefetch(xpt + (size_t)bb[q] * G4 + j0 + jj[q], 0, 1);
    }
    wait_asynccnt0();
    __syncthreads();

    v8f acc0 = {}; v8f acc1 = {};
#pragma unroll 2
    for (int kk = 0; kk < HSZ; kk += 32) {
      v16bf a  = ld16(hrow + kk + ahi, hrow + kk + ahi + 8);
      v16bf b0 = ld16(b0row + kk, b0row + kk + 8);
      v16bf b1 = ld16(b1row + kk, b1row + kk + 8);
      acc0 = wmma_bf16(a, b0, acc0);   // shared A frag across both gate tiles
      acc1 = wmma_bf16(a, b1, acc1);
    }
#pragma unroll
    for (int v = 0; v < 8; v++) {
      g_sh[gp       * 1024 + (rbase + v) * 16 + n] = acc0[v];
      g_sh[(gp + 2) * 1024 + (rbase + v) * 16 + n] = acc1[v];
    }
    __syncthreads();

#pragma unroll
    for (int q = 0; q < 4; q++) {
      int b = bb[q], j = jj[q];
      const float* xrow = xpt + (size_t)b * G4 + j0 + j;
      float gi = g_sh[0 * 1024 + b * 16 + j] + xrow[0];
      float gf = g_sh[1 * 1024 + b * 16 + j] + xrow[1024];
      float gg = g_sh[2 * 1024 + b * 16 + j] + xrow[2048];
      float go = g_sh[3 * 1024 + b * 16 + j] + xrow[3072];
      gi = sigmoidf_fast(gi);
      gf = sigmoidf_fast(gf);
      gg = tanhf(gg);
      go = sigmoidf_fast(go);
      float cn = gf * c_reg[q] + gi * gg;
      float hn = go * tanhf(cn);
      c_reg[q] = cn;
      out[(size_t)t * (BATCH * HSZ) + b * HSZ + j0 + j] = hn;
      hwrite[b * HSZ + j0 + j] = (bf16_t)hn;
      if (t == T_STEPS - 1) {
        size_t toff = (size_t)T_STEPS * BATCH * HSZ;
        out[toff + b * HSZ + j0 + j] = hn;                 // h_T
        out[toff + BATCH * HSZ + b * HSZ + j0 + j] = cn;   // c_T
      }
    }

    // grid-wide barrier (monotonic counter; 64 co-resident blocks)
    __threadfence();
    __syncthreads();
    if (tid == 0) {
      __hip_atomic_fetch_add(counter, 1u, __ATOMIC_ACQ_REL,
                             __HIP_MEMORY_SCOPE_AGENT);
      const unsigned goal = (unsigned)NBLK * (unsigned)(t + 1);
      while (__hip_atomic_load(counter, __ATOMIC_ACQUIRE,
                               __HIP_MEMORY_SCOPE_AGENT) < goal) {
        __builtin_amdgcn_s_sleep(1);
      }
    }
    __syncthreads();
  }
}

// ---------------------------------------------------------------------------
extern "C" void kernel_launch(void* const* d_in, const int* in_sizes, int n_in,
                              void* d_out, int out_size, void* d_ws,
                              size_t ws_size, hipStream_t stream) {
  (void)in_sizes; (void)n_in; (void)out_size; (void)ws_size;
  const float* x    = (const float*)d_in[0];
  const float* W_ih = (const float*)d_in[1];
  const float* W_hh = (const float*)d_in[2];
  const float* b_ih = (const float*)d_in[3];
  const float* b_hh = (const float*)d_in[4];
  const float* h0   = (const float*)d_in[5];
  const float* c0   = (const float*)d_in[6];
  float* out = (float*)d_out;

  char* ws = (char*)d_ws;
  unsigned* counter = (unsigned*)ws;                       // 256 B slot
  float* xp = (float*)(ws + 256);                          // [T*B, 4H] f32
  const size_t xp_bytes = (size_t)T_STEPS * BATCH * G4 * sizeof(float);
  bf16_t* hbuf = (bf16_t*)(ws + 256 + xp_bytes);           // 2x [B, H] bf16

  const size_t lds_bytes =
      2 * 64 * WSTRIDE * sizeof(bf16_t) + 4 * 64 * 16 * sizeof(float);
  (void)hipFuncSetAttribute((const void*)lstm_recurrent,
                            hipFuncAttributeMaxDynamicSharedMemorySize,
                            (int)lds_bytes);

  lstm_init<<<(BATCH * HSZ + 255) / 256, 256, 0, stream>>>(h0, hbuf, counter);

  dim3 g1(G4 / 64, (T_STEPS * BATCH) / 64);                // 64 x 1024 blocks
  gemm_xproj<<<g1, 256, 0, stream>>>(x, W_ih, b_ih, b_hh, xp);

  lstm_recurrent<<<NBLK, 256, lds_bytes, stream>>>(W_hh, xp, c0, out, hbuf,
                                                   counter);
}
